// MultiInSizeLinear_4801773437021
// MI455X (gfx1250) — matrix operand order repaired and checked by reading, not compiled, not run on hardware
//
#include <hip/hip_runtime.h>
#include <hip/hip_bf16.h>

typedef __attribute__((ext_vector_type(16))) __bf16   v16bf;
typedef __attribute__((ext_vector_type(8)))  float    v8f;
typedef __attribute__((ext_vector_type(8)))  unsigned v8u;

#define N_TOKENS  65536
#define MAXF      512
#define OUTF      1024
#define NGROUP    4
#define NWELEM    (NGROUP * OUTF * MAXF)   // 2,097,152 weight elements
#define MTILE     32                       // tokens per workgroup
#define PERM_CAP  (N_TOKENS + NGROUP * MTILE)

// d_ws layout (bytes):
//   [0 .. 128)                 : header ints: cnt[4], cursor[4], pbase[5]
//   [128 .. 128+PERM_CAP*4)    : perm (token index per padded slot, -1 = dummy)
//   [WHI_OFF ..)               : weight hi-bf16 bits (ushort, NWELEM)
//   [WLO_OFF ..)               : weight lo-bf16 bits (ushort, NWELEM)
#define PERM_OFF  128
#define WHI_OFF   (PERM_OFF + PERM_CAP * 4)          // 262784, 16B aligned
#define WLO_OFF   (WHI_OFF + NWELEM * 2)

__device__ __forceinline__ int size_to_group(int s) {
  return (s >= 512) ? 3 : (s >= 256) ? 2 : (s >= 128) ? 1 : 0;
}

// round-to-nearest-even f32 -> bf16 bit pattern
__device__ __forceinline__ unsigned bf16_rne_bits(float f) {
  unsigned u = __float_as_uint(f);
  return (u + (0x7fffu + ((u >> 16) & 1u))) >> 16;
}

// split f -> (hi bits, lo bits), hi = bf16_rne(f), lo = bf16_rne(f - hi)
__device__ __forceinline__ void split_bits(float f, unsigned short& h, unsigned short& l) {
  unsigned u  = __float_as_uint(f);
  unsigned hb = (u + (0x7fffu + ((u >> 16) & 1u))) & 0xffff0000u;
  float    lo = f - __uint_as_float(hb);
  h = (unsigned short)(hb >> 16);
  l = (unsigned short)bf16_rne_bits(lo);
}

// load WMMA operand fragment: 8 contiguous bf16 at p, 8 more at p+16 elements
// (matches both the 16-bit A 16x32 layout (per lane-half) and, with stride 16,
//  the B 32x16 layout, same K order as A: 8h+[0..8) and 16+8h+[0..8))
__device__ __forceinline__ v16bf ld_frag_gap(const unsigned short* p, int gap) {
  uint4 t0 = *(const uint4*)(p);
  uint4 t1 = *(const uint4*)(p + gap);
  v8u u = { t0.x, t0.y, t0.z, t0.w, t1.x, t1.y, t1.z, t1.w };
  return __builtin_bit_cast(v16bf, u);
}

// ---------------- setup kernels ----------------

__global__ void mis_init(int* hdr, int* perm) {
  int i = blockIdx.x * blockDim.x + threadIdx.x;
  if (i < 32) hdr[i] = 0;
  if (i < PERM_CAP) perm[i] = -1;
}

__global__ void mis_hist(const int* __restrict__ sz, int* hdr) {
  int i = blockIdx.x * blockDim.x + threadIdx.x;
  if (i < N_TOKENS) atomicAdd(&hdr[size_to_group(sz[i])], 1);
}

__global__ void mis_scan(int* hdr) {
  // single thread: padded (MTILE-aligned) segment bases
  int acc = 0;
  for (int g = 0; g < NGROUP; ++g) {
    hdr[8 + g] = acc;
    acc += ((hdr[g] + (MTILE - 1)) / MTILE) * MTILE;
  }
  hdr[12] = acc;
}

__global__ void mis_scatter(const int* __restrict__ sz, int* hdr, int* perm) {
  int i = blockIdx.x * blockDim.x + threadIdx.x;
  if (i < N_TOKENS) {
    int g   = size_to_group(sz[i]);
    int pos = atomicAdd(&hdr[4 + g], 1);
    perm[hdr[8 + g] + pos] = i;
  }
}

__global__ void mis_wsplit(const float* __restrict__ w,
                           unsigned short* __restrict__ whi,
                           unsigned short* __restrict__ wlo) {
  int i = blockIdx.x * blockDim.x + threadIdx.x;
  if (i < NWELEM) {
    unsigned short h, l;
    split_bits(w[i], h, l);
    whi[i] = h;
    wlo[i] = l;
  }
}

// ---------------- main GEMM kernel ----------------
// 512 threads = 16 waves. Block handles MTILE=32 tokens of ONE group.
// Wave w covers output columns [w*64, w*64+64): 2 M-subtiles x 4 N-subtiles.
// LDS: two separate bf16 planes (hi, lo) -> A fragments load with zero swizzle ALU.

__global__ __launch_bounds__(512)
void mis_gemm(const float* __restrict__ x,
              const float* __restrict__ bias,
              const unsigned short* __restrict__ whi,
              const unsigned short* __restrict__ wlo,
              const int* __restrict__ hdr,
              const int* __restrict__ perm,
              float* __restrict__ out) {
  __shared__ unsigned short lAhi[MTILE * MAXF];   // 32 KB
  __shared__ unsigned short lAlo[MTILE * MAXF];   // 32 KB  (total exactly 64 KB)

  const int tid = threadIdx.x;
  const int pb1 = hdr[9], pb2 = hdr[10], pb3 = hdr[11], pb4 = hdr[12];
  const int r0  = blockIdx.x * MTILE;
  if (r0 >= pb4) return;

  const int g  = (r0 >= pb3) ? 3 : (r0 >= pb2) ? 2 : (r0 >= pb1) ? 1 : 0;
  const int kg = 64 << g;                 // K for this group

  // ---- stage A tile: gather token rows, split to bf16 hi/lo planes ----
  {
    const int rsh    = 4 + g;             // log2(float4 per row) = log2(kg/4)
    const int total4 = (kg >> 2) * MTILE;
    for (int i4 = tid; i4 < total4; i4 += 512) {
      const int row  = i4 >> rsh;
      const int col4 = i4 - (row << rsh);
      const int tk   = perm[r0 + row];
      float4 v = make_float4(0.f, 0.f, 0.f, 0.f);
      if (tk >= 0) v = ((const float4*)(x + (size_t)tk * MAXF))[col4];
      ushort4 qh, ql;
      split_bits(v.x, qh.x, ql.x);
      split_bits(v.y, qh.y, ql.y);
      split_bits(v.z, qh.z, ql.z);
      split_bits(v.w, qh.w, ql.w);
      const int base = row * MAXF + (col4 << 2);
      *(ushort4*)&lAhi[base] = qh;
      *(ushort4*)&lAlo[base] = ql;
    }
  }
  __syncthreads();

  const int lane = tid & 31;
  const int wave = tid >> 5;      // 0..15
  const int ln16 = lane & 15;
  const int hlf  = lane >> 4;     // 0|1: K-half selector per ISA layouts
  const int n0   = wave << 6;     // 64 output columns per wave

  const unsigned short* wgh = whi + (size_t)g * OUTF * MAXF;
  const unsigned short* wgl = wlo + (size_t)g * OUTF * MAXF;

  v8f cacc[2][4];
  const v8f vzero = {0.f, 0.f, 0.f, 0.f, 0.f, 0.f, 0.f, 0.f};
#pragma unroll
  for (int mt = 0; mt < 2; ++mt)
#pragma unroll
    for (int j = 0; j < 4; ++j) cacc[mt][j] = vzero;

  for (int k = 0; k < kg; k += 32) {
    // A fragments for both 16-row subtiles.
    // 16-bit A 16x32 layout: lane-half h holds K = k+8h+[0..8) in elems 0..7
    // and K = k+16+8h+[0..8) in elems 8..15  -> two ds_load_b128, gap 16.
    v16bf Ah[2], Al[2];
#pragma unroll
    for (int mt = 0; mt < 2; ++mt) {
      const int aoff = ((mt << 4) + ln16) * MAXF + k + (hlf << 3);
      Ah[mt] = ld_frag_gap(&lAhi[aoff], 16);
      Al[mt] = ld_frag_gap(&lAlo[aoff], 16);
    }

#pragma unroll
    for (int j = 0; j < 4; ++j) {
      const int col = n0 + (j << 4) + ln16;
      // B 32x16 layout: lane-half h holds K = k+8h+[0..8) and k+16+8h+[0..8)
      const size_t wofs = (size_t)col * MAXF + (size_t)(k + (hlf << 3));
      v16bf Bh = ld_frag_gap(wgh + wofs, 16);
      v16bf Bl = ld_frag_gap(wgl + wofs, 16);
#pragma unroll
      for (int mt = 0; mt < 2; ++mt) {
        v8f a = cacc[mt][j];
        a = __builtin_amdgcn_wmma_f32_16x16x32_bf16(false, Ah[mt], false, Bh,
                                                    (short)0, a, false, false);
        a = __builtin_amdgcn_wmma_f32_16x16x32_bf16(false, Al[mt], false, Bh,
                                                    (short)0, a, false, false);
        a = __builtin_amdgcn_wmma_f32_16x16x32_bf16(false, Ah[mt], false, Bl,
                                                    (short)0, a, false, false);
        cacc[mt][j] = a;
      }
    }
  }

  // ---- epilogue: + bias, scatter rows to out (C layout: M = r + 8*hlf) ----
  const float* bg = bias + g * OUTF;
#pragma unroll
  for (int j = 0; j < 4; ++j) {
    const int col = n0 + (j << 4) + ln16;
    const float bv = bg[col];
#pragma unroll
    for (int mt = 0; mt < 2; ++mt) {
#pragma unroll
      for (int r = 0; r < 8; ++r) {
        const int tk = perm[r0 + (mt << 4) + r + (hlf << 3)];
        if (tk >= 0) out[(size_t)tk * OUTF + col] = cacc[mt][j][r] + bv;
      }
    }
  }
}

// ---------------- launcher ----------------

extern "C" void kernel_launch(void* const* d_in, const int* in_sizes, int n_in,
                              void* d_out, int out_size, void* d_ws, size_t ws_size,
                              hipStream_t stream) {
  const float* x  = (const float*)d_in[0];
  const int*   sz = (const int*)d_in[1];
  const float* w  = (const float*)d_in[2];
  const float* b  = (const float*)d_in[3];
  float*       o  = (float*)d_out;

  int* hdr  = (int*)d_ws;
  int* perm = (int*)((char*)d_ws + PERM_OFF);
  unsigned short* whi = (unsigned short*)((char*)d_ws + WHI_OFF);
  unsigned short* wlo = (unsigned short*)((char*)d_ws + WLO_OFF);

  mis_init   <<<(PERM_CAP + 255) / 256, 256, 0, stream>>>(hdr, perm);
  mis_hist   <<<(N_TOKENS + 255) / 256, 256, 0, stream>>>(sz, hdr);
  mis_scan   <<<1, 1, 0, stream>>>(hdr);
  mis_scatter<<<(N_TOKENS + 255) / 256, 256, 0, stream>>>(sz, hdr, perm);
  mis_wsplit <<<(NWELEM + 255) / 256, 256, 0, stream>>>(w, whi, wlo);

  const int gemm_blocks = (N_TOKENS + NGROUP * (MTILE - 1)) / MTILE + 1; // covers padded total
  mis_gemm<<<gemm_blocks, 512, 0, stream>>>(x, b, whi, wlo, hdr, perm, o);
}
